// GCNTopK_9139690406049
// MI455X (gfx1250) — matrix operand order repaired and checked by reading, not compiled, not run on hardware
//
#include <hip/hip_runtime.h>

// ---------------------------------------------------------------------------
// GCN + TopK pooling network for MI455X (gfx1250), wave32 + WMMA bf16.
// Input flattening assumption (setup_inputs() dict insertion order):
//   [0] x (32,1024,128) f32
//   [1] edge_index (32,16384,2) i32
//   [2 + 15*L + j] layer L leaves, j in {t_W1,t_b1,t_g1,t_be1,t_m1,t_v1,
//                                        t_W2,t_b2,t_g2,t_be2,t_m2,t_v2,
//                                        c_W,c_b,p_w}
//   [62..67] pred {W1,b1,W2,b2,W3,b3}
// Output: (32,4) f32.
// GEMMs run as bf16 x bf16 -> f32 accumulate via v_wmma_f32_16x16x32_bf16.
// ---------------------------------------------------------------------------

#define NB   32        // batch (graphs)
#define NN0  1024      // max nodes per graph (buffer stride)
#define NE   16384     // edges per graph
#define NF   128       // feature width
#define EPSV 1e-5f

typedef __attribute__((ext_vector_type(16))) __bf16 v16bf;
typedef __attribute__((ext_vector_type(8)))  float  v8f;

union FragBF { v16bf v; uint4 q[2]; };

__device__ __forceinline__ unsigned short f2bf(float f) {
  unsigned int u = __float_as_uint(f);
  u += 0x7FFFu + ((u >> 16) & 1u);          // round-to-nearest-even
  return (unsigned short)(u >> 16);
}

// --------------------------- staging / packing ------------------------------

// Convert first n rows per graph of a [NB][NN0][NF] f32 buffer to bf16.
__global__ void k_conv_bf16(const float* __restrict__ X,
                            unsigned short* __restrict__ Xbf, int n) {
  int idx = blockIdx.x * blockDim.x + threadIdx.x;
  if (idx >= NB * n * NF) return;
  int c = idx % NF;
  int r = (idx / NF) % n;
  int g = idx / (n * NF);
  size_t off = ((size_t)g * NN0 + r) * NF + c;
  Xbf[off] = f2bf(X[off]);
}

// Pack 128x128 f32 row-major W into WMMA dense-B fragment layout:
// Wpk[kt][nt][lane][j], lane<16: N=nt*16+lane, K=kt*32 + (lane>>4)*8 + (j&7) + (j>>3)*16.
__global__ void k_pack_w(const float* __restrict__ W,
                         unsigned short* __restrict__ Wpk) {
  int idx = blockIdx.x * blockDim.x + threadIdx.x;
  if (idx >= 4 * 8 * 32 * 16) return;
  int j    = idx & 15;
  int lane = (idx >> 4) & 31;
  int nt   = (idx >> 9) & 7;
  int kt   = idx >> 12;
  int ncol = nt * 16 + (lane & 15);
  int krow = kt * 32 + (lane >> 4) * 8 + (j & 7) + (j >> 3) * 16;
  Wpk[idx] = f2bf(W[krow * NF + ncol]);
}

// ------------------------------- WMMA GEMM ----------------------------------
// Out[g][row][col] = sum_k A[g][row][k] * W[k][col]   (+ epilogue)
// mode 0: plain; mode 2: bias + batchnorm + relu.
// One wave computes one 16-row x 128-col tile. Block = 4 waves.
__global__ void k_gemm_bf16(const unsigned short* __restrict__ Abf,
                            const unsigned short* __restrict__ Wpk,
                            float* __restrict__ Out, int n, int mode,
                            const float* __restrict__ bias,
                            const float* __restrict__ gam,
                            const float* __restrict__ bet,
                            const float* __restrict__ mu,
                            const float* __restrict__ var) {
  int lane = threadIdx.x & 31;
  int wave = threadIdx.x >> 5;
  int tiles = (n + 15) >> 4;
  int gw = blockIdx.x * 4 + wave;
  if (gw >= NB * tiles) return;          // whole-wave exit: EXEC stays full
  int g  = gw / tiles;
  int mt = gw % tiles;

  v8f acc[8];
#pragma unroll
  for (int i = 0; i < 8; ++i) {
    v8f z = {0.f, 0.f, 0.f, 0.f, 0.f, 0.f, 0.f, 0.f};
    acc[i] = z;
  }

  // A fragment addressing per ISA 16-bit A 16x32 layout:
  // lane<16 -> K in [kb..kb+7] and [kb+16..kb+23] with kb=0; lane>=16 -> kb=8.
  int m  = mt * 16 + (lane & 15);
  int kb = (lane >> 4) * 8;
  const unsigned short* arow = Abf + ((size_t)g * NN0 + m) * NF;

#pragma unroll
  for (int kt = 0; kt < 4; ++kt) {
    FragBF a;
    a.q[0] = *(const uint4*)(arow + kt * 32 + kb);
    a.q[1] = *(const uint4*)(arow + kt * 32 + kb + 16);
    if (kt < 3) __builtin_prefetch(arow + (kt + 1) * 32, 0, 0);
#pragma unroll
    for (int nt = 0; nt < 8; ++nt) {
      FragBF b;
      const unsigned short* bp = Wpk + (((kt * 8 + nt) * 32 + lane) * 16);
      b.q[0] = *(const uint4*)(bp);
      b.q[1] = *(const uint4*)(bp + 8);
      acc[nt] = __builtin_amdgcn_wmma_f32_16x16x32_bf16(
          false, a.v, false, b.v, (short)0, acc[nt], false, false);
    }
  }

  // C/D layout: VGPR v, lane<16 -> (M=v, N=lane); lane>=16 -> (M=8+v, N=lane-16)
  int rowBase = mt * 16 + ((lane >> 4) * 8);
  int colLane = lane & 15;
#pragma unroll
  for (int nt = 0; nt < 8; ++nt) {
    int col = nt * 16 + colLane;
    float bcol = 0.f, gc = 1.f, bec = 0.f, mc = 0.f, rstd = 1.f;
    if (mode == 2) {
      bcol = bias[col];
      gc   = gam[col];
      bec  = bet[col];
      mc   = mu[col];
      rstd = rsqrtf(var[col] + EPSV);
    }
#pragma unroll
    for (int v = 0; v < 8; ++v) {
      int row = rowBase + v;
      if (row < n) {
        float val = acc[nt][v];
        if (mode == 2) {
          val = (val + bcol - mc) * rstd * gc + bec;
          val = fmaxf(val, 0.f);
        }
        Out[((size_t)g * NN0 + row) * NF + col] = val;
      }
    }
  }
}

// ----------------------------- GCN aggregation ------------------------------

__global__ void k_init_ones(int* __restrict__ m, int count) {
  int idx = blockIdx.x * blockDim.x + threadIdx.x;
  if (idx < count) m[idx] = 1;
}
__global__ void k_init_zero_f(float* __restrict__ p, int count) {
  int idx = blockIdx.x * blockDim.x + threadIdx.x;
  if (idx < count) p[idx] = 0.f;
}
__global__ void k_deg_init(float* __restrict__ deg) {
  int idx = blockIdx.x * blockDim.x + threadIdx.x;
  if (idx < NB * NN0) deg[idx] = 2.0f;
}
__global__ void k_deg_scatter(const int* __restrict__ edges,
                              const int* __restrict__ mask,
                              float* __restrict__ deg) {
  int idx = blockIdx.x * blockDim.x + threadIdx.x;
  if (idx >= NB * NE) return;
  if (!mask[idx]) return;
  int g = idx / NE;
  int d = edges[idx * 2 + 1];
  atomicAdd(&deg[g * NN0 + d], 1.0f);
}
// agg = xw * (2/deg) + c_b   (self-loop term + bias folded)
__global__ void k_agg_init(const float* __restrict__ xw,
                           const float* __restrict__ deg,
                           const float* __restrict__ cb,
                           float* __restrict__ agg, int n) {
  int idx = blockIdx.x * blockDim.x + threadIdx.x;
  if (idx >= NB * n * NF) return;
  int c = idx % NF;
  int r = (idx / NF) % n;
  int g = idx / (n * NF);
  size_t off = ((size_t)g * NN0 + r) * NF + c;
  agg[off] = xw[off] * (2.0f / deg[g * NN0 + r]) + cb[c];
}
__global__ void k_agg_scatter(const int* __restrict__ edges,
                              const int* __restrict__ mask,
                              const float* __restrict__ xw,
                              const float* __restrict__ deg,
                              float* __restrict__ agg) {
  int idx = blockIdx.x * blockDim.x + threadIdx.x;
  if (idx >= NB * NE * NF) return;
  int c = idx % NF;
  int e = (idx / NF) % NE;
  int g = idx / (NE * NF);
  if (!mask[g * NE + e]) return;
  int s = edges[(g * NE + e) * 2 + 0];
  int d = edges[(g * NE + e) * 2 + 1];
  float coef = rsqrtf(deg[g * NN0 + s]) * rsqrtf(deg[g * NN0 + d]);
  atomicAdd(&agg[((size_t)g * NN0 + d) * NF + c],
            xw[((size_t)g * NN0 + s) * NF + c] * coef);
}

// ------------------------------ TopK pooling --------------------------------

__global__ void k_score(const float* __restrict__ H,
                        const float* __restrict__ pw,
                        float* __restrict__ score, int n) {
  int idx = blockIdx.x * blockDim.x + threadIdx.x;
  if (idx >= NB * n) return;
  int g = idx / n, r = idx % n;
  float s2 = 0.f, dot = 0.f;
  const float* row = H + ((size_t)g * NN0 + r) * NF;
  for (int c = 0; c < NF; ++c) {
    float w = pw[c];
    s2 += w * w;
    dot += row[c] * w;
  }
  score[g * NN0 + r] = tanhf(dot * rsqrtf(s2));
}

// Exact jax.lax.top_k ordering: descending, ties -> lower index first.
__global__ void k_topk(const float* __restrict__ score,
                       float* __restrict__ vals, int* __restrict__ perm,
                       int* __restrict__ inv, int n, int k) {
  int g = blockIdx.x;
  int t = threadIdx.x;
  __shared__ float ss[NN0];
  ss[t] = (t < n) ? score[g * NN0 + t] : -3.4e38f;
  __syncthreads();
  if (t < n) {
    float s = ss[t];
    int rank = 0;
    for (int j = 0; j < n; ++j) {
      float sj = ss[j];
      rank += (sj > s) || (sj == s && j < t);
    }
    if (rank < k) {
      vals[g * NN0 + rank] = s;
      perm[g * NN0 + rank] = t;
    }
    inv[g * NN0 + t] = (rank < k) ? rank : -1;
  }
}

__global__ void k_gather(const float* __restrict__ H,
                         const float* __restrict__ vals,
                         const int* __restrict__ perm,
                         float* __restrict__ Xn, int k) {
  int idx = blockIdx.x * blockDim.x + threadIdx.x;
  if (idx >= NB * k * NF) return;
  int c = idx % NF;
  int r = (idx / NF) % k;
  int g = idx / (k * NF);
  int src = perm[g * NN0 + r];
  Xn[((size_t)g * NN0 + r) * NF + c] =
      H[((size_t)g * NN0 + src) * NF + c] * vals[g * NN0 + r];
}

__global__ void k_edge_remap(const int* __restrict__ eIn,
                             const int* __restrict__ mIn,
                             const int* __restrict__ inv,
                             int* __restrict__ eOut, int* __restrict__ mOut) {
  int idx = blockIdx.x * blockDim.x + threadIdx.x;
  if (idx >= NB * NE) return;
  int g = idx / NE;
  int s = eIn[idx * 2 + 0];
  int d = eIn[idx * 2 + 1];
  int ns = inv[g * NN0 + s];
  int nd = inv[g * NN0 + d];
  int mk = (mIn[idx] != 0) && (ns >= 0) && (nd >= 0);
  eOut[idx * 2 + 0] = mk ? ns : 0;
  eOut[idx * 2 + 1] = mk ? nd : 0;
  mOut[idx] = mk;
}

// readout += concat(max over nodes, mean over nodes)
__global__ void k_readout(const float* __restrict__ H,
                          float* __restrict__ racc, int k) {
  int idx = blockIdx.x * blockDim.x + threadIdx.x;
  if (idx >= NB * NF) return;
  int g = idx / NF, c = idx % NF;
  float mx = -3.4e38f, sm = 0.f;
  for (int r = 0; r < k; ++r) {
    float v = H[((size_t)g * NN0 + r) * NF + c];
    mx = fmaxf(mx, v);
    sm += v;
  }
  racc[g * 256 + c] += mx;
  racc[g * 256 + 128 + c] += sm / (float)k;
}

// ------------------------------- predictor ----------------------------------

__global__ void k_predictor(const float* __restrict__ racc,
                            const float* __restrict__ W1, const float* __restrict__ b1,
                            const float* __restrict__ W2, const float* __restrict__ b2,
                            const float* __restrict__ W3, const float* __restrict__ b3,
                            float* __restrict__ out) {
  int g = blockIdx.x, t = threadIdx.x;
  __shared__ float r[256];
  __shared__ float h1[128];
  __shared__ float h2[64];
  r[t]       = racc[g * 256 + t] * 0.25f;
  r[128 + t] = racc[g * 256 + 128 + t] * 0.25f;
  __syncthreads();
  float a = b1[t];
  for (int j = 0; j < 256; ++j) a += r[j] * W1[j * 128 + t];
  h1[t] = fmaxf(a, 0.f);
  __syncthreads();
  if (t < 64) {
    float a2 = b2[t];
    for (int j = 0; j < 128; ++j) a2 += h1[j] * W2[j * 64 + t];
    h2[t] = fmaxf(a2, 0.f);
  }
  __syncthreads();
  if (t < 4) {
    float a3 = b3[t];
    for (int j = 0; j < 64; ++j) a3 += h2[j] * W3[j * 4 + t];
    out[g * 4 + t] = a3;
  }
}

// ------------------------------- launcher -----------------------------------

#define CEILDIV(a, b) (((a) + (b) - 1) / (b))

extern "C" void kernel_launch(void* const* d_in, const int* in_sizes, int n_in,
                              void* d_out, int out_size, void* d_ws, size_t ws_size,
                              hipStream_t stream) {
  (void)in_sizes; (void)n_in; (void)out_size; (void)ws_size;
  const float* x0 = (const float*)d_in[0];
  const int*   e0 = (const int*)d_in[1];
  const float* pW1 = (const float*)d_in[62];
  const float* pb1 = (const float*)d_in[63];
  const float* pW2 = (const float*)d_in[64];
  const float* pb2 = (const float*)d_in[65];
  const float* pW3 = (const float*)d_in[66];
  const float* pb3 = (const float*)d_in[67];

  char* ws = (char*)d_ws;
  size_t off = 0;
  auto alloc = [&](size_t bytes) -> void* {
    off = (off + 255) & ~(size_t)255;
    void* p = ws + off;
    off += bytes;
    return p;
  };
  const size_t nodeBytesF = (size_t)NB * NN0 * NF * sizeof(float);
  float* bufA  = (float*)alloc(nodeBytesF);
  float* bufB  = (float*)alloc(nodeBytesF);
  float* bufT  = (float*)alloc(nodeBytesF);     // MLP temp / AGG
  float* bufXW = (float*)alloc(nodeBytesF);
  unsigned short* bufBF = (unsigned short*)alloc((size_t)NB * NN0 * NF * 2);
  unsigned short* wpk   = (unsigned short*)alloc(NF * NF * 2);
  int* edgesA = (int*)alloc((size_t)NB * NE * 2 * sizeof(int));
  int* edgesB = (int*)alloc((size_t)NB * NE * 2 * sizeof(int));
  int* maskA  = (int*)alloc((size_t)NB * NE * sizeof(int));
  int* maskB  = (int*)alloc((size_t)NB * NE * sizeof(int));
  float* deg  = (float*)alloc((size_t)NB * NN0 * sizeof(float));
  float* scr  = (float*)alloc((size_t)NB * NN0 * sizeof(float));
  float* vals = (float*)alloc((size_t)NB * NN0 * sizeof(float));
  int*   perm = (int*)alloc((size_t)NB * NN0 * sizeof(int));
  int*   inv  = (int*)alloc((size_t)NB * NN0 * sizeof(int));
  float* racc = (float*)alloc((size_t)NB * 256 * sizeof(float));

  k_init_ones<<<CEILDIV(NB * NE, 256), 256, 0, stream>>>(maskA, NB * NE);
  k_init_zero_f<<<CEILDIV(NB * 256, 256), 256, 0, stream>>>(racc, NB * 256);

  const float* Xcur = x0;
  float* Xnext = bufA;
  const int* eCur = e0;
  const int* mCur = maskA;
  int* eNext = edgesB;
  int* mNext = maskB;

  int n = NN0;
  for (int L = 0; L < 4; ++L) {
    int k = (4 * n + 4) / 5;                 // ceil(0.8 * n)
    const float* LW1 = (const float*)d_in[2 + 15 * L + 0];
    const float* Lb1 = (const float*)d_in[2 + 15 * L + 1];
    const float* Lg1 = (const float*)d_in[2 + 15 * L + 2];
    const float* Le1 = (const float*)d_in[2 + 15 * L + 3];
    const float* Lm1 = (const float*)d_in[2 + 15 * L + 4];
    const float* Lv1 = (const float*)d_in[2 + 15 * L + 5];
    const float* LW2 = (const float*)d_in[2 + 15 * L + 6];
    const float* Lb2 = (const float*)d_in[2 + 15 * L + 7];
    const float* Lg2 = (const float*)d_in[2 + 15 * L + 8];
    const float* Le2 = (const float*)d_in[2 + 15 * L + 9];
    const float* Lm2 = (const float*)d_in[2 + 15 * L + 10];
    const float* Lv2 = (const float*)d_in[2 + 15 * L + 11];
    const float* LcW = (const float*)d_in[2 + 15 * L + 12];
    const float* Lcb = (const float*)d_in[2 + 15 * L + 13];
    const float* Lpw = (const float*)d_in[2 + 15 * L + 14];

    int gemmBlocks = CEILDIV(NB * CEILDIV(n, 16), 4);
    int nElem = NB * n * NF;

    // h = relu(bn(x @ W1 + b1))
    k_conv_bf16<<<CEILDIV(nElem, 256), 256, 0, stream>>>(Xcur, bufBF, n);
    k_pack_w<<<CEILDIV(16384, 256), 256, 0, stream>>>(LW1, wpk);
    k_gemm_bf16<<<gemmBlocks, 128, 0, stream>>>(bufBF, wpk, bufT, n, 2,
                                                Lb1, Lg1, Le1, Lm1, Lv1);
    // h = relu(bn(h @ W2 + b2))
    k_conv_bf16<<<CEILDIV(nElem, 256), 256, 0, stream>>>(bufT, bufBF, n);
    k_pack_w<<<CEILDIV(16384, 256), 256, 0, stream>>>(LW2, wpk);
    k_gemm_bf16<<<gemmBlocks, 128, 0, stream>>>(bufBF, wpk, bufT, n, 2,
                                                Lb2, Lg2, Le2, Lm2, Lv2);
    // xw = h @ c_W
    k_conv_bf16<<<CEILDIV(nElem, 256), 256, 0, stream>>>(bufT, bufBF, n);
    k_pack_w<<<CEILDIV(16384, 256), 256, 0, stream>>>(LcW, wpk);
    k_gemm_bf16<<<gemmBlocks, 128, 0, stream>>>(bufBF, wpk, bufXW, n, 0,
                                                nullptr, nullptr, nullptr,
                                                nullptr, nullptr);
    // GCN aggregation (bufT becomes AGG)
    k_deg_init<<<CEILDIV(NB * NN0, 256), 256, 0, stream>>>(deg);
    k_deg_scatter<<<CEILDIV(NB * NE, 256), 256, 0, stream>>>(eCur, mCur, deg);
    k_agg_init<<<CEILDIV(nElem, 256), 256, 0, stream>>>(bufXW, deg, Lcb, bufT, n);
    k_agg_scatter<<<CEILDIV(NB * NE * NF, 256), 256, 0, stream>>>(eCur, mCur,
                                                                  bufXW, deg, bufT);
    // TopK pool
    k_score<<<CEILDIV(NB * n, 256), 256, 0, stream>>>(bufT, Lpw, scr, n);
    k_topk<<<NB, 1024, 0, stream>>>(scr, vals, perm, inv, n, k);
    k_gather<<<CEILDIV(NB * k * NF, 256), 256, 0, stream>>>(bufT, vals, perm,
                                                            Xnext, k);
    k_edge_remap<<<CEILDIV(NB * NE, 256), 256, 0, stream>>>(eCur, mCur, inv,
                                                            eNext, mNext);
    k_readout<<<CEILDIV(NB * NF, 256), 256, 0, stream>>>(Xnext, racc, k);

    // rotate buffers
    Xcur  = Xnext;
    Xnext = (Xcur == bufA) ? bufB : bufA;
    eCur  = eNext;
    mCur  = mNext;
    eNext = (eCur == edgesA) ? edgesB : edgesA;
    mNext = (mCur == maskA) ? maskB : maskA;
    n = k;
  }

  k_predictor<<<NB, 128, 0, stream>>>(racc, pW1, pb1, pW2, pb2, pW3, pb3,
                                      (float*)d_out);
}
